// BMPNodeBlock_38809324487018
// MI455X (gfx1250) — matrix operand order repaired and checked by reading, not compiled, not run on hardware
//
#include <hip/hip_runtime.h>
#include <hip/hip_bf16.h>
#include <math.h>

#define H 64
#define TWOH 128
#define ROWS_PER_BLOCK 128
#define WPB 8  // waves per block (wave32)

typedef __attribute__((ext_vector_type(16))) __bf16 v16bf;
typedef __attribute__((ext_vector_type(8)))  __bf16 v8bf;
typedef __attribute__((ext_vector_type(4)))  __bf16 v4bf;
typedef __attribute__((ext_vector_type(2)))  __bf16 v2bf;
typedef __attribute__((ext_vector_type(8)))  float  v8f;

// Monotone float -> u32 encoding so GLOBAL_ATOMIC_MAX_U32 == float max (exact, deterministic).
#define ENC_NEG_INF 0x007FFFFFu   // enc(-inf)
__device__ __forceinline__ unsigned encf(float f) {
  unsigned u = __float_as_uint(f);
  return (u & 0x80000000u) ? ~u : (u | 0x80000000u);
}
__device__ __forceinline__ float decf(unsigned e) {
  if (e == ENC_NEG_INF) return 0.0f;                 // empty segment -> 0 (reference semantics)
  unsigned u = (e & 0x80000000u) ? (e ^ 0x80000000u) : ~e;
  return __uint_as_float(u);
}
__device__ __forceinline__ v16bf join16(v8bf lo, v8bf hi) {
  union { v16bf v; v8bf h[2]; } u; u.h[0] = lo; u.h[1] = hi; return u.v;
}

// ---------------- K1: init scatter buffers to enc(-inf) ----------------
__global__ void k_init(unsigned* __restrict__ buf, long long total_enc) {
  long long i = (long long)blockIdx.x * blockDim.x + threadIdx.x;
  if (i < total_enc) buf[i] = ENC_NEG_INF;
}

// ---------------- K2: edge scatter-max (1 wave per edge) ----------------
__global__ void k_scatter(const int* __restrict__ ei, const float* __restrict__ msg,
                          unsigned* __restrict__ fwd, unsigned* __restrict__ bwd, int E) {
  int e = blockIdx.x * WPB + (threadIdx.x >> 5);
  if (e >= E) return;
  int L = threadIdx.x & 31;
  int c = ei[(long long)E + e];   // edge_index[1][e]
  int r = ei[e];                  // edge_index[0][e]
  long long mb = (long long)e * H;
  unsigned e0 = encf(msg[mb + L]);
  unsigned e1 = encf(msg[mb + L + 32]);
  atomicMax(&fwd[(long long)c * H + L],       e0);
  atomicMax(&fwd[(long long)c * H + L + 32],  e1);
  atomicMax(&bwd[(long long)r * H + L],       e0);
  atomicMax(&bwd[(long long)r * H + L + 32],  e1);
}

// ---------------- K3: y1 = concat(fwd,bwd) @ W1 + b1, + BN partial stats ----------------
__global__ void __launch_bounds__(256) k_gemm1(
    const unsigned* __restrict__ fwd, const unsigned* __restrict__ bwd,
    const float* __restrict__ W1, const float* __restrict__ b1,
    float* __restrict__ y, float* __restrict__ pb, int NN) {
  __shared__ __bf16 Wt[H][TWOH + 8];            // W1^T [n][k], padded vs bank conflicts
  __shared__ __bf16 As[WPB][16][TWOH + 8];      // per-wave 16x128 A tile
  __shared__ float  s_ps[WPB][H], s_pq[WPB][H]; // per-wave per-col partials (deterministic)

  const int tid  = threadIdx.x;
  const int w    = tid >> 5;
  const int lane = tid & 31;
  const int half = lane >> 4;
  const int l15  = lane & 15;
  const int rowBase = blockIdx.x * ROWS_PER_BLOCK + w * 16;
  const bool fullTile = (rowBase + 16) <= NN;   // wave-uniform: only the last block is partial

  // stage W1^T as bf16 (coalesced read, one-time transpose into LDS)
  for (int idx = tid; idx < TWOH * H; idx += 256) {
    int k = idx >> 6, n = idx & 63;
    Wt[n][k] = (__bf16)W1[idx];
  }
  // stage A: decode fwd (cols 0..63) / bwd (cols 64..127), zeros for tail rows
  const int cbase = (lane < 16) ? (4 * lane) : (4 * (lane - 16));
  const unsigned* src = (lane < 16) ? fwd : bwd;
  const int col = (lane < 16) ? cbase : (H + cbase);
  if (fullTile) {
    for (int r = 0; r < 16; ++r) {
      const unsigned* p = src + (long long)(rowBase + r) * H + cbase;
      *(v4bf*)&As[w][r][col] = (v4bf){(__bf16)decf(p[0]), (__bf16)decf(p[1]),
                                      (__bf16)decf(p[2]), (__bf16)decf(p[3])};
    }
  } else {
    for (int r = 0; r < 16; ++r) {
      int row = rowBase + r;
      float v0 = 0.f, v1 = 0.f, v2 = 0.f, v3 = 0.f;
      if (row < NN) {
        const unsigned* p = src + (long long)row * H + cbase;
        v0 = decf(p[0]); v1 = decf(p[1]); v2 = decf(p[2]); v3 = decf(p[3]);
      }
      *(v4bf*)&As[w][r][col] = (v4bf){(__bf16)v0, (__bf16)v1, (__bf16)v2, (__bf16)v3};
    }
  }
  __syncthreads();

  const __bf16* ap = &As[w][l15][0];
  for (int nt = 0; nt < 4; ++nt) {
    const int n = nt * 16 + l15;
    float bias = b1[n];
    v8f c;
    for (int i = 0; i < 8; ++i) c[i] = bias;
    const __bf16* bp = &Wt[n][0];
    for (int kb = 0; kb < TWOH; kb += 32) {
      // A 16x32 bf16: lane half selects K {0..7,16..23} vs {8..15,24..31}
      v16bf a = join16(*(const v8bf*)(ap + kb + 8 * half),
                       *(const v8bf*)(ap + kb + 16 + 8 * half));
      // B 32x16 bf16: lane = column, half selects K {0..7,16..23} vs {8..15,24..31}
      v16bf b = join16(*(const v8bf*)(bp + kb + 8 * half),
                       *(const v8bf*)(bp + kb + 16 + 8 * half));
      c = __builtin_amdgcn_wmma_f32_16x16x32_bf16(false, a, false, b, (short)0, c, false, false);
    }
    // write y1 + accumulate per-column stats; branch once per tile, not per element
    float ps = 0.f, pq = 0.f;
    float* yp = y + (long long)(rowBase + half * 8) * H + n;   // C layout: M = i + 8*half
    if (fullTile) {
      for (int i = 0; i < 8; ++i) {
        float v = c[i];
        yp[(long long)i * H] = v;
        ps += v; pq += v * v;
      }
    } else {
      for (int i = 0; i < 8; ++i) {
        if (rowBase + half * 8 + i < NN) {
          float v = c[i];
          yp[(long long)i * H] = v;
          ps += v; pq += v * v;
        }
      }
    }
    ps += __shfl_xor(ps, 16, 32);           // lanes L and L+16 share column n
    pq += __shfl_xor(pq, 16, 32);
    if (half == 0) { s_ps[w][n] = ps; s_pq[w][n] = pq; }
  }
  __syncthreads();
  if (tid < H) {                            // deterministic per-block partial
    float S = 0.f, Q = 0.f;
    for (int ww = 0; ww < WPB; ++ww) { S += s_ps[ww][tid]; Q += s_pq[ww][tid]; }
    pb[(long long)blockIdx.x * 2 * H + tid]     = S;
    pb[(long long)blockIdx.x * 2 * H + H + tid] = Q;
  }
}

// ---------------- K4/K6: fold partials -> BN scale/shift ----------------
__global__ void k_finalize(const float* __restrict__ pb, int nblk,
                           const float* __restrict__ g, const float* __restrict__ be,
                           float* __restrict__ scale, float* __restrict__ shift, float invN) {
  int t = threadIdx.x;
  if (t < H) {
    float S = 0.f, Q = 0.f;
    for (int b = 0; b < nblk; ++b) {
      S += pb[(long long)b * 2 * H + t];
      Q += pb[(long long)b * 2 * H + H + t];
    }
    float m = S * invN;
    float var = Q * invN - m * m;
    float r = rsqrtf(var + 1e-5f);
    float s = g[t] * r;
    scale[t] = s;
    shift[t] = be[t] - m * s;
  }
}

// ---------------- K5: h1 = relu(bn(y1)); y2 = h1 @ W2 + b2 (in place) + stats ----------------
__global__ void __launch_bounds__(256) k_gemm2(
    const float* __restrict__ W2, const float* __restrict__ b2,
    const float* __restrict__ sc1, const float* __restrict__ sh1,
    float* __restrict__ y, float* __restrict__ pb, int NN) {
  __shared__ __bf16 Wt[H][H + 8];
  __shared__ __bf16 As[WPB][16][H + 8];
  __shared__ float  s_ps[WPB][H], s_pq[WPB][H];

  const int tid  = threadIdx.x;
  const int w    = tid >> 5;
  const int lane = tid & 31;
  const int half = lane >> 4;
  const int l15  = lane & 15;
  const int rowBase = blockIdx.x * ROWS_PER_BLOCK + w * 16;
  const bool fullTile = (rowBase + 16) <= NN;

  for (int idx = tid; idx < H * H; idx += 256) {
    int k = idx >> 6, n = idx & 63;
    Wt[n][k] = (__bf16)W2[idx];
  }
  const int c0 = 2 * lane, c1 = 2 * lane + 1;
  const float a0s = sc1[c0], a0b = sh1[c0], a1s = sc1[c1], a1b = sh1[c1];
  if (fullTile) {
    for (int r = 0; r < 16; ++r) {
      const float* p = y + (long long)(rowBase + r) * H;
      float v0 = fmaxf(p[c0] * a0s + a0b, 0.f);   // BN + ReLU fused into staging
      float v1 = fmaxf(p[c1] * a1s + a1b, 0.f);
      *(v2bf*)&As[w][r][c0] = (v2bf){(__bf16)v0, (__bf16)v1};
    }
  } else {
    for (int r = 0; r < 16; ++r) {
      int row = rowBase + r;
      float v0 = 0.f, v1 = 0.f;
      if (row < NN) {
        const float* p = y + (long long)row * H;
        v0 = fmaxf(p[c0] * a0s + a0b, 0.f);
        v1 = fmaxf(p[c1] * a1s + a1b, 0.f);
      }
      *(v2bf*)&As[w][r][c0] = (v2bf){(__bf16)v0, (__bf16)v1};
    }
  }
  __syncthreads();

  const __bf16* ap = &As[w][l15][0];
  for (int nt = 0; nt < 4; ++nt) {
    const int n = nt * 16 + l15;
    float bias = b2[n];
    v8f c;
    for (int i = 0; i < 8; ++i) c[i] = bias;
    const __bf16* bp = &Wt[n][0];
    for (int kb = 0; kb < H; kb += 32) {
      v16bf a = join16(*(const v8bf*)(ap + kb + 8 * half),
                       *(const v8bf*)(ap + kb + 16 + 8 * half));
      v16bf b = join16(*(const v8bf*)(bp + kb + 8 * half),
                       *(const v8bf*)(bp + kb + 16 + 8 * half));
      c = __builtin_amdgcn_wmma_f32_16x16x32_bf16(false, a, false, b, (short)0, c, false, false);
    }
    float ps = 0.f, pq = 0.f;
    float* yp = y + (long long)(rowBase + half * 8) * H + n;
    if (fullTile) {
      for (int i = 0; i < 8; ++i) {
        float v = c[i];
        yp[(long long)i * H] = v;           // in place: all reads staged before barrier
        ps += v; pq += v * v;
      }
    } else {
      for (int i = 0; i < 8; ++i) {
        if (rowBase + half * 8 + i < NN) {
          float v = c[i];
          yp[(long long)i * H] = v;
          ps += v; pq += v * v;
        }
      }
    }
    ps += __shfl_xor(ps, 16, 32);
    pq += __shfl_xor(pq, 16, 32);
    if (half == 0) { s_ps[w][n] = ps; s_pq[w][n] = pq; }
  }
  __syncthreads();
  if (tid < H) {
    float S = 0.f, Q = 0.f;
    for (int ww = 0; ww < WPB; ++ww) { S += s_ps[ww][tid]; Q += s_pq[ww][tid]; }
    pb[(long long)blockIdx.x * 2 * H + tid]     = S;
    pb[(long long)blockIdx.x * 2 * H + H + tid] = Q;
  }
}

// ---------------- K7: h2 = relu(bn(y2)) in place; att = sigmoid(h2 @ Wa + ba) ----------------
__global__ void k_bn2_att(const float* __restrict__ sc2, const float* __restrict__ sh2,
                          const float* __restrict__ Wa, const float* __restrict__ ba,
                          float* __restrict__ y, float* __restrict__ att, int NN) {
  int row = blockIdx.x * WPB + (threadIdx.x >> 5);
  if (row >= NN) return;
  int L = threadIdx.x & 31;
  long long base = (long long)row * H;
  float v0 = fmaxf(y[base + L]      * sc2[L]      + sh2[L],      0.f);
  float v1 = fmaxf(y[base + L + 32] * sc2[L + 32] + sh2[L + 32], 0.f);
  y[base + L]      = v0;
  y[base + L + 32] = v1;
  float z = v0 * Wa[L] + v1 * Wa[L + 32];
  for (int off = 16; off; off >>= 1) z += __shfl_xor(z, off, 32);
  if (L == 0) att[row] = 1.0f / (1.0f + expf(-(z + ba[0])));
}

extern "C" void kernel_launch(void* const* d_in, const int* in_sizes, int n_in,
                              void* d_out, int out_size, void* d_ws, size_t ws_size,
                              hipStream_t stream) {
  (void)n_in; (void)out_size; (void)ws_size;
  const int*   ei  = (const int*)d_in[1];
  const float* msg = (const float*)d_in[2];
  const float* W1  = (const float*)d_in[3];
  const float* b1  = (const float*)d_in[4];
  const float* g1  = (const float*)d_in[5];
  const float* be1 = (const float*)d_in[6];
  const float* W2  = (const float*)d_in[7];
  const float* b2  = (const float*)d_in[8];
  const float* g2  = (const float*)d_in[9];
  const float* be2 = (const float*)d_in[10];
  const float* Wa  = (const float*)d_in[11];
  const float* ba  = (const float*)d_in[12];

  const int NN = in_sizes[0] / H;   // nodes (x only contributes its shape)
  const int E  = in_sizes[2] / H;   // edges
  const int gblk = (NN + ROWS_PER_BLOCK - 1) / ROWS_PER_BLOCK;

  // workspace layout
  unsigned* fwd = (unsigned*)d_ws;                       // [NN,64] enc
  unsigned* bwd = fwd + (size_t)NN * H;                  // [NN,64] enc
  float* pb1 = (float*)(bwd + (size_t)NN * H);           // [gblk,128] partials (sum|sumsq)
  float* pb2 = pb1 + (size_t)gblk * 2 * H;               // [gblk,128]
  float* sc1 = pb2 + (size_t)gblk * 2 * H;
  float* sh1 = sc1 + H;
  float* sc2 = sh1 + H;
  float* sh2 = sc2 + H;

  float* y   = (float*)d_out;                            // [NN,64]: y1 -> y2 -> h2
  float* att = y + (size_t)NN * H;                       // [NN]

  long long total_enc = (long long)NN * TWOH;
  k_init<<<(unsigned)((total_enc + 255) / 256), 256, 0, stream>>>((unsigned*)d_ws, total_enc);
  k_scatter<<<(E + WPB - 1) / WPB, 256, 0, stream>>>(ei, msg, fwd, bwd, E);
  k_gemm1<<<gblk, 256, 0, stream>>>(fwd, bwd, W1, b1, y, pb1, NN);
  k_finalize<<<1, 64, 0, stream>>>(pb1, gblk, g1, be1, sc1, sh1, 1.0f / (float)NN);
  k_gemm2<<<gblk, 256, 0, stream>>>(W2, b2, sc1, sh1, y, pb2, NN);
  k_finalize<<<1, 64, 0, stream>>>(pb2, gblk, g2, be2, sc2, sh2, 1.0f / (float)NN);
  k_bn2_att<<<(NN + WPB - 1) / WPB, 256, 0, stream>>>(sc2, sh2, Wa, ba, y, att, NN);
}
